// CNNTransformerCriticBN_11235634446685
// MI455X (gfx1250) — matrix operand order repaired and checked by reading, not compiled, not run on hardware
//
#include <hip/hip_runtime.h>
#include <cstddef>

// ---------------------------------------------------------------------------
// CDNA5 / gfx1250 WMMA types and fragment helpers (wave32)
// ---------------------------------------------------------------------------
typedef __attribute__((ext_vector_type(16))) _Float16 v16h;
typedef __attribute__((ext_vector_type(8)))  float    v8f;

union AFrag { v16h v; _Float16 h[16]; unsigned u[8]; };

#define LRELU_SLOPE 0.01f

// A matrix fragment (16x32 f16, row-major source A[M,K], lda in halves).
// ISA layout: lane holds row m=lane&15; VGPR r holds K pair:
//   K = (r&4?16:0) + (lane&16?8:0) + (r&3)*2 (+k0)
static __device__ __forceinline__ v16h wmma_load_a(const _Float16* __restrict__ A,
                                                   int lda, int k0, int lane) {
  AFrag f;
  const _Float16* row = A + (size_t)(lane & 15) * lda + k0 + ((lane & 16) ? 8 : 0);
#pragma unroll
  for (int r = 0; r < 8; ++r)
    f.u[r] = *(const unsigned*)(row + ((r & 4) ? 16 : 0) + ((r & 3) << 1));
  return f.v;
}

// B matrix fragment (32x16 f16) sourced from W[N,K] row-major (i.e. B = W^T).
// Layout (same K order as A): lane holds col n=lane&15;
//   K = (r&4?16:0) + (lane&16?8:0) + (r&3)*2 (+k0)
static __device__ __forceinline__ v16h wmma_load_b(const _Float16* __restrict__ W,
                                                   int ldw, int k0, int lane) {
  AFrag f;
  const _Float16* row = W + (size_t)(lane & 15) * ldw + k0 + ((lane & 16) ? 8 : 0);
#pragma unroll
  for (int r = 0; r < 8; ++r)
    f.u[r] = *(const unsigned*)(row + ((r & 4) ? 16 : 0) + ((r & 3) << 1));
  return f.v;
}

// ---------------------------------------------------------------------------
// Direct 2x2 valid conv, f32 (small fraction of total FLOPs)
// in [N,IC,IH,IW] -> out [N,OC,OH,OW];  out = conv(in)*in_scale + bias
// ---------------------------------------------------------------------------
__global__ __launch_bounds__(256) void conv2x2_kernel(
    const float* __restrict__ in, const float* __restrict__ w,
    const float* __restrict__ bias, float* __restrict__ out,
    int NIMG, int IC, int IH, int IW, int OC, int OH, int OW, float in_scale) {
  int idx = blockIdx.x * 256 + threadIdx.x;
  int total = NIMG * OC * OH * OW;
  if (idx >= total) return;
  int ox = idx % OW;
  int oy = (idx / OW) % OH;
  int oc = (idx / (OW * OH)) % OC;
  int n  = idx / (OW * OH * OC);
  const float* ib = in + (size_t)n * IC * IH * IW;
  const float* wb = w + (size_t)oc * IC * 4;
  float acc = 0.f;
  for (int ic = 0; ic < IC; ++ic) {
    const float* p = ib + ic * IH * IW + oy * IW + ox;
    const float* q = wb + ic * 4;
    acc += p[0] * q[0] + p[1] * q[1] + p[IW] * q[2] + p[IW + 1] * q[3];
  }
  out[idx] = acc * in_scale + bias[oc];
}

// ---------------------------------------------------------------------------
// Training-mode BatchNorm stats: one block per channel; writes fused
// scale = g*rsqrt(var+eps), shift = b - mean*scale
// ---------------------------------------------------------------------------
__global__ __launch_bounds__(256) void bn_stats_kernel(
    const float* __restrict__ y, float* __restrict__ scale, float* __restrict__ shift,
    const float* __restrict__ g, const float* __restrict__ bta,
    int C, int HW, int NIMG) {
  __shared__ float ssum[256];
  __shared__ float ssq[256];
  int c = blockIdx.x;
  int t = threadIdx.x;
  float s = 0.f, s2 = 0.f;
  int per = NIMG * HW;
  for (int i = t; i < per; i += 256) {
    int n = i / HW, sp = i - n * HW;
    float v = y[((size_t)n * C + c) * HW + sp];
    s += v; s2 += v * v;
  }
  ssum[t] = s; ssq[t] = s2;
  __syncthreads();
  for (int off = 128; off > 0; off >>= 1) {
    if (t < off) { ssum[t] += ssum[t + off]; ssq[t] += ssq[t + off]; }
    __syncthreads();
  }
  if (t == 0) {
    float inv = 1.f / (float)per;
    float m = ssum[0] * inv;
    float var = ssq[0] * inv - m * m;
    float sc = g[c] * rsqrtf(var + 1e-5f);
    scale[c] = sc;
    shift[c] = bta[c] - m * sc;
  }
}

__global__ __launch_bounds__(256) void bn_relu_kernel(
    float* __restrict__ y, const float* __restrict__ scale,
    const float* __restrict__ shift, int C, int HW, int total) {
  int idx = blockIdx.x * 256 + threadIdx.x;
  if (idx >= total) return;
  int c = (idx / HW) % C;
  float v = fmaf(y[idx], scale[c], shift[c]);
  y[idx] = v > 0.f ? v : 0.f;
}

// ---------------------------------------------------------------------------
// f32 -> f16 conversion (activations and weights)
// ---------------------------------------------------------------------------
__global__ __launch_bounds__(256) void cvt_f16_kernel(
    const float* __restrict__ in, _Float16* __restrict__ out, int n) {
  int idx = blockIdx.x * 256 + threadIdx.x;
  if (idx < n) out[idx] = (_Float16)in[idx];
}

// sap_w [128,133] -> f16 padded [128,160] (K padded with zeros to x32)
__global__ __launch_bounds__(256) void cvt_sapw_kernel(
    const float* __restrict__ w, _Float16* __restrict__ out) {
  int idx = blockIdx.x * 256 + threadIdx.x;
  if (idx >= 128 * 160) return;
  int k = idx % 160, n = idx / 160;
  out[idx] = (k < 133) ? (_Float16)w[n * 133 + k] : (_Float16)0.f;
}

// xa/xp [8192,160] f16 = cat(states[8192,128], act[8192,5], zero pad)
__global__ __launch_bounds__(256) void build_cat_kernel(
    const _Float16* __restrict__ states, const float* __restrict__ act,
    _Float16* __restrict__ out, int M) {
  int idx = blockIdx.x * 256 + threadIdx.x;
  if (idx >= M * 160) return;
  int k = idx % 160, m = idx / 160;
  _Float16 v = (_Float16)0.f;
  if (k < 128)      v = states[(size_t)m * 128 + k];
  else if (k < 133) v = (_Float16)act[(size_t)m * 5 + (k - 128)];
  out[idx] = v;
}

// ---------------------------------------------------------------------------
// Strip-mined WMMA GEMM: out[M,N](f16) = act(A[M,K](f16) @ W[N,K]^T + bias)
// One wave per 16x64 output strip (4 accumulators): the A fragment is loaded
// once per K-step and amortized over 4 WMMAs -> 4x fewer A-side loads and
// denser v_wmma issue. Requires N % 64 == 0, K % 32 == 0.
// ---------------------------------------------------------------------------
__global__ __launch_bounds__(256) void gemm_wmma_f16(
    const _Float16* __restrict__ A, const _Float16* __restrict__ W,
    const float* __restrict__ bias, _Float16* __restrict__ out,
    int M, int N, int K, int actf) {
  int wave = (blockIdx.x * 256 + threadIdx.x) >> 5;
  int lane = threadIdx.x & 31;
  int nstrips = N >> 6;              // strips of 64 columns
  int tm = wave / nstrips;
  int ts = wave - tm * nstrips;
  if (tm * 16 >= M) return;          // wave-uniform: EXEC stays all-ones for WMMA
  const _Float16* Ab = A + (size_t)tm * 16 * K;
  const _Float16* Wb = W + (size_t)ts * 64 * K;
  v8f c0 = {}, c1 = {}, c2 = {}, c3 = {};
  for (int k0 = 0; k0 < K; k0 += 32) {
    v16h a  = wmma_load_a(Ab, K, k0, lane);
    v16h b0 = wmma_load_b(Wb + 0 * 16 * K, K, k0, lane);
    v16h b1 = wmma_load_b(Wb + 1 * 16 * K, K, k0, lane);
    v16h b2 = wmma_load_b(Wb + 2 * 16 * K, K, k0, lane);
    v16h b3 = wmma_load_b(Wb + 3 * 16 * K, K, k0, lane);
    c0 = __builtin_amdgcn_wmma_f32_16x16x32_f16(false, a, false, b0, (short)0, c0, false, false);
    c1 = __builtin_amdgcn_wmma_f32_16x16x32_f16(false, a, false, b1, (short)0, c1, false, false);
    c2 = __builtin_amdgcn_wmma_f32_16x16x32_f16(false, a, false, b2, (short)0, c2, false, false);
    c3 = __builtin_amdgcn_wmma_f32_16x16x32_f16(false, a, false, b3, (short)0, c3, false, false);
  }
  int rbase = tm * 16 + ((lane & 16) ? 8 : 0);
  v8f acc[4] = {c0, c1, c2, c3};
#pragma unroll
  for (int t = 0; t < 4; ++t) {
    int col = ts * 64 + t * 16 + (lane & 15);
    float bv = bias ? bias[col] : 0.f;
#pragma unroll
    for (int r = 0; r < 8; ++r) {
      float y = acc[t][r] + bv;
      if (actf) y = (y >= 0.f) ? y : LRELU_SLOPE * y;
      out[(size_t)(rbase + r) * N + col] = (_Float16)y;
    }
  }
}

// ---------------------------------------------------------------------------
// Attention: one wave per batch b. S = q[b] @ k[b]^T / sqrt(128),
// row-softmax, write f32 ret_weight and f16 copy for node mixing.
// ---------------------------------------------------------------------------
__global__ __launch_bounds__(256) void attention_kernel(
    const _Float16* __restrict__ q, const _Float16* __restrict__ k,
    float* __restrict__ w_out, _Float16* __restrict__ w16) {
  int b    = (blockIdx.x * 256 + threadIdx.x) >> 5;  // 0..511 exact
  int lane = threadIdx.x & 31;
  const _Float16* qb = q + (size_t)b * 16 * 128;
  const _Float16* kb = k + (size_t)b * 16 * 128;
  v8f c = {};
  for (int k0 = 0; k0 < 128; k0 += 32) {
    v16h a  = wmma_load_a(qb, 128, k0, lane);
    v16h bf = wmma_load_b(kb, 128, k0, lane);
    c = __builtin_amdgcn_wmma_f32_16x16x32_f16(false, a, false, bf, (short)0, c,
                                               false, false);
  }
  const float sc = 0.08838834764831845f;  // 1/sqrt(128)
  int col = lane & 15;
  int rb  = (lane & 16) ? 8 : 0;
#pragma unroll
  for (int r = 0; r < 8; ++r) {
    float x = c[r] * sc;
    // row softmax: each row lives in a 16-lane half -> xor masks < 16
    float mx = x;
    for (int msk = 1; msk < 16; msk <<= 1) mx = fmaxf(mx, __shfl_xor(mx, msk, 32));
    float ex = __expf(x - mx);
    float sm = ex;
    for (int msk = 1; msk < 16; msk <<= 1) sm += __shfl_xor(sm, msk, 32);
    float wv = ex / sm;
    size_t o = (size_t)b * 256 + (size_t)(rb + r) * 16 + col;
    w_out[o] = wv;
    w16[o]   = (_Float16)wv;
  }
}

// ---------------------------------------------------------------------------
// Fused node-einsum + fv1 + fv2: one wave per (b, j).
//   node[i,:] = sum_k weight[b,i,k] * Av_j[k,:]   (Av_j row k = av_p if k==j else av_a)
//   v = lrelu(node @ fv1_w^T); Value[b,i,j] = v @ fv2_w
// node tile is staged through LDS to re-fragment it as a WMMA A-operand.
// ---------------------------------------------------------------------------
__global__ __launch_bounds__(256) void node_value_kernel(
    const _Float16* __restrict__ wgt16, const _Float16* __restrict__ ava,
    const _Float16* __restrict__ avp, const _Float16* __restrict__ fv1w,
    const float* __restrict__ fv2w, float* __restrict__ val_out) {
  __shared__ __align__(16) _Float16 smem[8 * 16 * 128];
  int wid  = threadIdx.x >> 5;
  int lane = threadIdx.x & 31;
  int gw   = blockIdx.x * 8 + wid;  // 0..8191 exact
  int b = gw >> 4, j = gw & 15;
  _Float16* nb = smem + wid * (16 * 128);

  // A fragment from weight[b] (16x16, K padded 16->32 with zeros)
  AFrag af;
  {
    const _Float16* row = wgt16 + (size_t)b * 256 + (lane & 15) * 16 + ((lane & 16) ? 8 : 0);
#pragma unroll
    for (int r = 0; r < 4; ++r) af.u[r] = *(const unsigned*)(row + (r << 1));
#pragma unroll
    for (int r = 4; r < 8; ++r) af.u[r] = 0u;
  }

  const int dcol = lane & 15;
  const int hi   = (lane & 16) ? 1 : 0;  // hi lane-half holds K=8..15; K=16..31 (r>=4) -> zero
  const int rb   = hi ? 8 : 0;

#pragma unroll
  for (int t = 0; t < 8; ++t) {
    AFrag bf;
    int d = t * 16 + dcol;
#pragma unroll
    for (int r = 0; r < 8; ++r) {
      int ka = 8 * hi + 2 * (r & 3), kb2 = ka + 1;
      _Float16 va = ((ka  == j) ? avp : ava)[((size_t)(b * 16 + ka))  * 128 + d];
      _Float16 vb = ((kb2 == j) ? avp : ava)[((size_t)(b * 16 + kb2)) * 128 + d];
      bf.h[2 * r]     = (r >= 4) ? (_Float16)0.f : va;
      bf.h[2 * r + 1] = (r >= 4) ? (_Float16)0.f : vb;
    }
    v8f c = {};
    c = __builtin_amdgcn_wmma_f32_16x16x32_f16(false, af.v, false, bf.v, (short)0,
                                               c, false, false);
#pragma unroll
    for (int r = 0; r < 8; ++r)
      nb[(size_t)(rb + r) * 128 + t * 16 + dcol] = (_Float16)c[r];
  }
  asm volatile("s_wait_dscnt 0" ::: "memory");  // node tile visible before re-load
  __syncthreads();

  // fv1: v[16,64] = node[16,128] @ fv1_w[64,128]^T
  v8f cv[4] = {};
#pragma unroll
  for (int s = 0; s < 4; ++s) {
    int k0 = 32 * s;
    v16h a = wmma_load_a(nb, 128, k0, lane);
#pragma unroll
    for (int tn = 0; tn < 4; ++tn) {
      v16h bb = wmma_load_b(fv1w + (size_t)tn * 16 * 128, 128, k0, lane);
      cv[tn] = __builtin_amdgcn_wmma_f32_16x16x32_f16(false, a, false, bb,
                                                      (short)0, cv[tn], false, false);
    }
  }

  // lrelu + fv2 dot along d (cross-lane reduce within 16-lane halves)
  float rowsum[8] = {0.f, 0.f, 0.f, 0.f, 0.f, 0.f, 0.f, 0.f};
#pragma unroll
  for (int tn = 0; tn < 4; ++tn) {
    float w2 = fv2w[tn * 16 + dcol];
#pragma unroll
    for (int r = 0; r < 8; ++r) {
      float y = cv[tn][r];
      y = (y >= 0.f) ? y : LRELU_SLOPE * y;
      rowsum[r] += y * w2;
    }
  }
#pragma unroll
  for (int r = 0; r < 8; ++r)
    for (int msk = 1; msk < 16; msk <<= 1)
      rowsum[r] += __shfl_xor(rowsum[r], msk, 32);
  if (dcol == 0) {
#pragma unroll
    for (int r = 0; r < 8; ++r)
      val_out[(size_t)b * 256 + (size_t)(rb + r) * 16 + j] = rowsum[r];
  }
}

// ---------------------------------------------------------------------------
// Host side
// ---------------------------------------------------------------------------
static inline int cdiv_i(int a, int b) { return (a + b - 1) / b; }

extern "C" void kernel_launch(void* const* d_in, const int* in_sizes, int n_in,
                              void* d_out, int out_size, void* d_ws, size_t ws_size,
                              hipStream_t stream) {
  (void)in_sizes; (void)n_in; (void)out_size; (void)ws_size;
  const float* img  = (const float*)d_in[0];
  const float* pol  = (const float*)d_in[1];
  const float* actn = (const float*)d_in[2];
  const float* c1w = (const float*)d_in[3];  const float* c1b = (const float*)d_in[4];
  const float* g1  = (const float*)d_in[5];  const float* bb1 = (const float*)d_in[6];
  const float* c2w = (const float*)d_in[7];  const float* c2b = (const float*)d_in[8];
  const float* g2  = (const float*)d_in[9];  const float* bb2 = (const float*)d_in[10];
  const float* c3w = (const float*)d_in[11]; const float* c3b = (const float*)d_in[12];
  const float* g3  = (const float*)d_in[13]; const float* bb3 = (const float*)d_in[14];
  const float* p1w = (const float*)d_in[15]; const float* p1b = (const float*)d_in[16];
  const float* p2w = (const float*)d_in[17]; const float* p2b = (const float*)d_in[18];
  const float* p3w = (const float*)d_in[19]; const float* p3b = (const float*)d_in[20];
  const float* sew = (const float*)d_in[21]; const float* seb = (const float*)d_in[22];
  const float* kw  = (const float*)d_in[23]; const float* qw  = (const float*)d_in[24];
  const float* spw = (const float*)d_in[25]; const float* spb = (const float*)d_in[26];
  const float* avw = (const float*)d_in[27];
  const float* f1w = (const float*)d_in[28]; const float* f2w = (const float*)d_in[29];

  const int NIMG = 8192;  // B*N = 512*16
  char* ws = (char*)d_ws;

  // --- workspace map (aliased; peak ~90.2 MB) ---
  const size_t OFF_Y2    = 37748736;   // y1 region size
  const size_t OFF_STATS = OFF_Y2 + 52428800;
  float* y1 = (float*)(ws);                // [8192,32,6,6]  (also reused for y3 + smalls)
  float* y2 = (float*)(ws + OFF_Y2);       // [8192,64,5,5]  (also reused for emb16)
  float* y3 = (float*)(ws);                // [8192,64,4,4]  (aliases y1; x1 dead)
  _Float16* emb = (_Float16*)(ws + OFF_Y2);        // [8192,1024] (aliases y2; x2 dead)
  float* scale = (float*)(ws + OFF_STATS);         // [64]
  float* shift = (float*)(ws + OFF_STATS) + 64;    // [64]
  // smalls inside y1 region (valid once convs complete)
  _Float16* h1     = (_Float16*)(ws + 0);
  _Float16* h2     = (_Float16*)(ws + 8388608);
  _Float16* states = (_Float16*)(ws + 10485760);
  _Float16* se     = (_Float16*)(ws + 12582912);
  _Float16* kbuf   = (_Float16*)(ws + 14680064);
  _Float16* qbuf   = (_Float16*)(ws + 16777216);
  _Float16* xa     = (_Float16*)(ws + 18874368);
  _Float16* xp     = (_Float16*)(ws + 21495808);
  _Float16* ea     = (_Float16*)(ws + 24117248);
  _Float16* ep     = (_Float16*)(ws + 26214400);
  _Float16* ava    = (_Float16*)(ws + 28311552);
  _Float16* avp    = (_Float16*)(ws + 30408704);
  _Float16* wgt16  = (_Float16*)(ws + 32505856);
  _Float16* p1w16  = (_Float16*)(ws + 32768000);
  _Float16* p2w16  = (_Float16*)(ws + 33816576);
  _Float16* p3w16  = (_Float16*)(ws + 33947648);
  _Float16* sew16  = (_Float16*)(ws + 33980416);
  _Float16* keyw16 = (_Float16*)(ws + 34013184);
  _Float16* qryw16 = (_Float16*)(ws + 34045952);
  _Float16* sapw16 = (_Float16*)(ws + 34078720);
  _Float16* avw16  = (_Float16*)(ws + 34119680);
  _Float16* f1w16  = (_Float16*)(ws + 34152448);

  float* out_val = (float*)d_out;            // Value [512,16,16,1]
  float* out_wgt = out_val + 512 * 256;      // ret_weight [512,16,16]

  // ---- conv stack (f32 direct conv + two-pass BN + ReLU) ----
  {
    int tot = NIMG * 32 * 36;
    conv2x2_kernel<<<cdiv_i(tot, 256), 256, 0, stream>>>(
        img, c1w, c1b, y1, NIMG, 3, 7, 7, 32, 6, 6, 1.f / 255.f);
    bn_stats_kernel<<<32, 256, 0, stream>>>(y1, scale, shift, g1, bb1, 32, 36, NIMG);
    bn_relu_kernel<<<cdiv_i(tot, 256), 256, 0, stream>>>(y1, scale, shift, 32, 36, tot);
  }
  {
    int tot = NIMG * 64 * 25;
    conv2x2_kernel<<<cdiv_i(tot, 256), 256, 0, stream>>>(
        y1, c2w, c2b, y2, NIMG, 32, 6, 6, 64, 5, 5, 1.f);
    bn_stats_kernel<<<64, 256, 0, stream>>>(y2, scale, shift, g2, bb2, 64, 25, NIMG);
    bn_relu_kernel<<<cdiv_i(tot, 256), 256, 0, stream>>>(y2, scale, shift, 64, 25, tot);
  }
  {
    int tot = NIMG * 64 * 16;
    conv2x2_kernel<<<cdiv_i(tot, 256), 256, 0, stream>>>(
        y2, c3w, c3b, y3, NIMG, 64, 5, 5, 64, 4, 4, 1.f);
    bn_stats_kernel<<<64, 256, 0, stream>>>(y3, scale, shift, g3, bb3, 64, 16, NIMG);
    bn_relu_kernel<<<cdiv_i(tot, 256), 256, 0, stream>>>(y3, scale, shift, 64, 16, tot);
    cvt_f16_kernel<<<cdiv_i(tot, 256), 256, 0, stream>>>(y3, emb, tot);  // emb [8192,1024]
  }

  // ---- weight conversions (y3 now dead; smalls region is free) ----
  cvt_f16_kernel<<<cdiv_i(512 * 1024, 256), 256, 0, stream>>>(p1w, p1w16, 512 * 1024);
  cvt_f16_kernel<<<cdiv_i(128 * 512, 256), 256, 0, stream>>>(p2w, p2w16, 128 * 512);
  cvt_f16_kernel<<<cdiv_i(128 * 128, 256), 256, 0, stream>>>(p3w, p3w16, 128 * 128);
  cvt_f16_kernel<<<cdiv_i(128 * 128, 256), 256, 0, stream>>>(sew, sew16, 128 * 128);
  cvt_f16_kernel<<<cdiv_i(128 * 128, 256), 256, 0, stream>>>(kw, keyw16, 128 * 128);
  cvt_f16_kernel<<<cdiv_i(128 * 128, 256), 256, 0, stream>>>(qw, qryw16, 128 * 128);
  cvt_f16_kernel<<<cdiv_i(128 * 128, 256), 256, 0, stream>>>(avw, avw16, 128 * 128);
  cvt_f16_kernel<<<cdiv_i(64 * 128, 256), 256, 0, stream>>>(f1w, f1w16, 64 * 128);
  cvt_sapw_kernel<<<cdiv_i(128 * 160, 256), 256, 0, stream>>>(spw, sapw16);

  // ---- WMMA GEMM chain (one wave per 16x64 strip) ----
  auto gemm = [&](const _Float16* A, const _Float16* W, const float* bias,
                  _Float16* O, int M, int N, int K, int actf) {
    int waves = (M / 16) * (N / 64);
    gemm_wmma_f16<<<cdiv_i(waves * 32, 256), 256, 0, stream>>>(A, W, bias, O, M, N, K, actf);
  };
  gemm(emb, p1w16, p1b, h1, NIMG, 512, 1024, 1);   // post1 + lrelu
  gemm(h1, p2w16, p2b, h2, NIMG, 128, 512, 1);     // post2 + lrelu
  gemm(h2, p3w16, p3b, states, NIMG, 128, 128, 1); // post3 + lrelu -> states
  gemm(states, sew16, seb, se, NIMG, 128, 128, 1); // se + lrelu
  gemm(se, keyw16, nullptr, kbuf, NIMG, 128, 128, 0);
  gemm(se, qryw16, nullptr, qbuf, NIMG, 128, 128, 0);

  // ---- attention (one wave per batch; 512 waves exact) ----
  attention_kernel<<<64, 256, 0, stream>>>(qbuf, kbuf, out_wgt, wgt16);

  // ---- state-action embedding branch ----
  build_cat_kernel<<<cdiv_i(NIMG * 160, 256), 256, 0, stream>>>(states, actn, xa, NIMG);
  build_cat_kernel<<<cdiv_i(NIMG * 160, 256), 256, 0, stream>>>(states, pol, xp, NIMG);
  gemm(xa, sapw16, spb, ea, NIMG, 128, 160, 1);    // lrelu
  gemm(xp, sapw16, spb, ep, NIMG, 128, 160, 1);    // lrelu
  gemm(ea, avw16, nullptr, ava, NIMG, 128, 128, 0);
  gemm(ep, avw16, nullptr, avp, NIMG, 128, 128, 0);

  // ---- fused node-einsum + fv1 + fv2 (one wave per (b,j); 8192 waves) ----
  node_value_kernel<<<1024, 256, 0, stream>>>(wgt16, ava, avp, f1w16, f2w, out_val);
}
